// Combined_2branches_32770600468567
// MI455X (gfx1250) — matrix operand order repaired and checked by reading, not compiled, not run on hardware
//
#include <hip/hip_runtime.h>
#include <math.h>

typedef __attribute__((ext_vector_type(16))) _Float16 v16h;
typedef __attribute__((ext_vector_type(8)))  float    v8f;

#define N_ROWS 32768
#define NIMG   64
#define HDIM   128
#define KF     3
#define NBF    2
#define EPSBN  1e-5f
#define WPB    4   // waves per block (wave32)

// ---------------------------------------------------------------- zero out
__global__ void zero_kernel(float* o) {
  if (threadIdx.x == 0 && blockIdx.x == 0) o[0] = 0.f;
}

// ---------------------------------------------------------------- CNN branch
struct CnnP {
  const float *x, *c1W, *c1b, *g1, *b1, *m1, *v1;
  const float *c2W, *c2b, *g2, *b2, *m2, *v2;
  const float *f1W, *f1b, *pw, *g3, *b3, *m3, *v3;
  const float *f2W, *f2b;
  float* ctx_out;   // (64,128) in workspace
  int c_in, col_off;
};

__global__ void cnn_kernel(CnnP p) {
  __shared__ float xs[32 * 9];
  __shared__ float y1[16 * 9];
  __shared__ float y2[32 * 4];
  __shared__ float y3[32];
  __shared__ float h1[16];
  const int t = threadIdx.x;
  const int img = blockIdx.x;
  const int ni = p.c_in * 9;
  for (int k = t; k < ni; k += blockDim.x) xs[k] = p.x[img * ni + k];
  __syncthreads();
  // conv1 (2x2, pad bottom/right by 1 -> 3x3 out) + bn1 + relu
  if (t < 144) {
    int oc = t / 9, pos = t % 9, h = pos / 3, w = pos % 3;
    float s = p.c1b[oc];
    for (int ic = 0; ic < p.c_in; ++ic)
      for (int kh = 0; kh < 2; ++kh)
        for (int kw = 0; kw < 2; ++kw) {
          int ih = h + kh, iw = w + kw;
          if (ih < 3 && iw < 3)
            s += xs[ic * 9 + ih * 3 + iw] * p.c1W[((oc * p.c_in + ic) * 2 + kh) * 2 + kw];
        }
    s = (s - p.m1[oc]) * (p.g1[oc] * rsqrtf(p.v1[oc] + EPSBN)) + p.b1[oc];
    y1[t] = fmaxf(s, 0.f);
  }
  __syncthreads();
  // conv2 (2x2 VALID -> 2x2 out) + bn2 + relu
  if (t < 128) {
    int oc = t / 4, pos = t % 4, h = pos / 2, w = pos % 2;
    float s = p.c2b[oc];
    for (int ic = 0; ic < 16; ++ic)
      for (int kh = 0; kh < 2; ++kh)
        for (int kw = 0; kw < 2; ++kw)
          s += y1[ic * 9 + (h + kh) * 3 + (w + kw)] * p.c2W[((oc * 16 + ic) * 2 + kh) * 2 + kw];
    s = (s - p.m2[oc]) * (p.g2[oc] * rsqrtf(p.v2[oc] + EPSBN)) + p.b2[oc];
    y2[t] = fmaxf(s, 0.f);
  }
  __syncthreads();
  // 2x2 maxpool -> 1x1
  if (t < 32)
    y3[t] = fmaxf(fmaxf(y2[t * 4], y2[t * 4 + 1]), fmaxf(y2[t * 4 + 2], y2[t * 4 + 3]));
  __syncthreads();
  // fc1 + prelu + bn3
  if (t < 16) {
    float s = p.f1b[t];
    for (int j = 0; j < 32; ++j) s += y3[j] * p.f1W[t * 32 + j];
    s = (s >= 0.f) ? s : p.pw[0] * s;
    s = (s - p.m3[t]) * (p.g3[t] * rsqrtf(p.v3[t] + EPSBN)) + p.b3[t];
    h1[t] = s;
  }
  __syncthreads();
  // fc2 -> 64 ctx values
  if (t < 64) {
    float s = p.f2b[t];
    for (int j = 0; j < 16; ++j) s += h1[j] * p.f2W[t * 16 + j];
    p.ctx_out[img * HDIM + p.col_off + t] = s;
  }
}

// ------------------------------------------- hoisted ctx matmuls (64 unique rows)
struct CtxmmP {
  const float *ctx, *ctx_W, *ctx_b, *init_b, *C_W, *C_b;
  float *hc, *gate;
};
__global__ void ctxmm_kernel(CtxmmP p) {
  int idx = blockIdx.x * blockDim.x + threadIdx.x;
  if (idx >= KF * 3 * NIMG * HDIM) return;
  int c   = idx & (HDIM - 1);
  int img = (idx >> 7) & (NIMG - 1);
  int m   = (idx >> 13) % 3;
  int i   = (idx >> 13) / 3;
  const float* cv = p.ctx + img * HDIM;
  if (m == 0) {
    const float* w = p.ctx_W + ((size_t)i * HDIM + c) * HDIM;
    float s = p.ctx_b[i * HDIM + c] + p.init_b[i * HDIM + c];
    for (int k = 0; k < HDIM; ++k) s += cv[k] * w[k];
    p.hc[((size_t)i * NIMG + img) * HDIM + c] = s;
  } else {
    int b = m - 1;
    const float* w = p.C_W + ((size_t)(i * NBF + b) * HDIM + c) * HDIM;
    float s = p.C_b[(i * NBF + b) * HDIM + c];
    for (int k = 0; k < HDIM; ++k) s += cv[k] * w[k];
    p.gate[((size_t)(i * NBF + b) * NIMG + img) * HDIM + c] = 1.f / (1.f + expf(-s));
  }
}

// ------------------------------------ pack A_W/B_W into f16 WMMA B-fragment layout
// frag id bits: kt[1:0], n[4:2], which[5], b[6], i[8:7]; 512 halves per fragment.
// B (32x16, KxN) layout: lane = N (lane&15); element e -> K = kt*32 +
// (e<8 ? e : e+8) + (lane>=16?8:0).  Value = W[N][K].
struct PackP { const float* A_W; const float* B_W; _Float16* packW; };
__global__ void pack_kernel(PackP p) {
  int idx = blockIdx.x * blockDim.x + threadIdx.x;
  if (idx >= 384 * 512) return;
  int frag = idx >> 9, rem = idx & 511;
  int lam = rem >> 4, e = rem & 15;
  int kt = frag & 3, n = (frag >> 2) & 7, which = (frag >> 5) & 1;
  int b = (frag >> 6) & 1, i = frag >> 7;
  int N  = n * 16 + (lam & 15);
  int Kg = kt * 32 + ((lam & 16) ? 8 : 0) + (e < 8 ? e : e + 8);
  const float* W = (which == 0 ? p.A_W : p.B_W) + ((size_t)(i * NBF + b) * HDIM + N) * HDIM + Kg;
  p.packW[idx] = (_Float16)(*W);
}

// ---------------------------------------------------------------- fused flow
struct FlowP {
  const float *samples, *init_W, *A_b, *B_b, *fin_W, *fin_b;
  const float *lu_lower, *lu_upper, *lu_udiag, *lu_bias;
  const int* perms;
  const float *hc, *gate;
  const _Float16* packW;
  float* out;
};

__global__ __launch_bounds__(WPB * 32) void flow_kernel(FlowP p) {
  __shared__ float    h_s[WPB][16][HDIM];   // fp32 hidden state (residual stream)
  __shared__ _Float16 t_s[WPB][16][HDIM];   // f16 staging for second matmul A-source
  __shared__ float    z_s[WPB][16][2];
  __shared__ float    x_s[WPB][16][2];
  __shared__ float    zz_s[WPB][16][2];
  __shared__ float    out_s[WPB][16][4];
  __shared__ float    lq_s[WPB][16];

  const int wv = threadIdx.x >> 5;
  const int lane = threadIdx.x & 31;
  const int rowbase = blockIdx.x * (WPB * 16) + wv * 16;
  const int img = rowbase >> 9;            // 512 samples per image
  const int arow = lane & 15;
  const int rsel = (lane & 16) ? 8 : 0;    // D-fragment row-half select

  // warm the packed weights (gfx1250 global_prefetch_b8)
  __builtin_prefetch(p.packW + lane * 64, 0, 3);

  if (lane < 16) {
    int r = rowbase + lane;
    z_s[wv][lane][0] = p.samples[2 * r];
    z_s[wv][lane][1] = p.samples[2 * r + 1];
    lq_s[wv][lane] = 0.f;
  }
  __syncthreads();

  for (int i = KF - 1; i >= 0; --i) {
    int p0 = p.perms[2 * i], p1 = p.perms[2 * i + 1];
    float ud0 = log1pf(expf(p.lu_udiag[2 * i])) + 0.001f;       // softplus
    float ud1 = log1pf(expf(p.lu_udiag[2 * i + 1])) + 0.001f;
    float lo = p.lu_lower[i], up = p.lu_upper[i];
    float bi0 = p.lu_bias[2 * i], bi1 = p.lu_bias[2 * i + 1];
    if (lane < 16) {
      float zp0 = z_s[wv][lane][p0];
      float zp1 = z_s[wv][lane][p1];
      x_s[wv][lane][0] = zp0 * ud0 + zp1 * up + bi0;
      x_s[wv][lane][1] = zp0 * (lo * ud0) + zp1 * (lo * up + ud1) + bi1;
      zz_s[wv][lane][0] = 0.f;
      zz_s[wv][lane][1] = 0.f;
      lq_s[wv][lane] += logf(ud0) + logf(ud1);
    }
    __syncthreads();

    const float* hcL = p.hc + ((size_t)i * NIMG + img) * HDIM;
    const float* iwL = p.init_W + (size_t)i * HDIM * 2;
    float s0 = 1.f, s1 = 1.f;  // scales of last fixed-point iteration

    for (int iter = 0; iter < 2; ++iter) {
      // h = hc + zz @ init_W.T  (per-row broadcast of the hoisted ctx term)
      for (int idx = lane; idx < 16 * HDIM; idx += 32) {
        int r = idx >> 7, c = idx & (HDIM - 1);
        h_s[wv][r][c] = hcL[c] + zz_s[wv][r][0] * iwL[c * 2] + zz_s[wv][r][1] * iwL[c * 2 + 1];
      }
      __syncthreads();

      for (int b = 0; b < NBF; ++b) {
        const float* Ab = p.A_b + (size_t)(i * NBF + b) * HDIM;
        const float* Bb = p.B_b + (size_t)(i * NBF + b) * HDIM;
        const _Float16* wA = p.packW + (size_t)((i * NBF + b) * 2 + 0) * 32 * 512;
        const _Float16* wB = p.packW + (size_t)((i * NBF + b) * 2 + 1) * 32 * 512;
        const float* gateL = p.gate + ((size_t)(i * NBF + b) * NIMG + img) * HDIM;

        // ---- t = relu( relu(h) @ A.T + A_b ) via v_wmma_f32_16x16x32_f16
        v8f acc[8];
#pragma unroll
        for (int n = 0; n < 8; ++n) {
          float bv = Ab[n * 16 + arow];
#pragma unroll
          for (int v = 0; v < 8; ++v) acc[n][v] = bv;
        }
#pragma unroll
        for (int kt = 0; kt < 4; ++kt) {
          // A fragment: 16-bit A layout — K = kt*32 + (e<8 ? e : e+8) + (lane>=16 ? 8 : 0)
          v16h a;
          int kb = kt * 32 + ((lane & 16) ? 8 : 0);
#pragma unroll
          for (int e = 0; e < 8; ++e) {
            a[e]     = (_Float16)fmaxf(h_s[wv][arow][kb + e], 0.f);
            a[e + 8] = (_Float16)fmaxf(h_s[wv][arow][kb + 16 + e], 0.f);
          }
#pragma unroll
          for (int n = 0; n < 8; ++n) {
            v16h bf = *(const v16h*)(wA + (size_t)((n * 4 + kt) * 32 + lane) * 16);
            acc[n] = __builtin_amdgcn_wmma_f32_16x16x32_f16(
                false, a, false, bf, (short)0, acc[n], false, false);
          }
        }
        // stage relu(acc) to LDS in row-major f16 (D-fragment -> tile scatter)
#pragma unroll
        for (int n = 0; n < 8; ++n) {
          int col = n * 16 + arow;
#pragma unroll
          for (int v = 0; v < 8; ++v)
            t_s[wv][v + rsel][col] = (_Float16)fmaxf(acc[n][v], 0.f);
        }
        __syncthreads();

        // ---- t2 = t @ B.T + B_b ; h += t2 * gate
        v8f acc2[8];
#pragma unroll
        for (int n = 0; n < 8; ++n) {
          float bv = Bb[n * 16 + arow];
#pragma unroll
          for (int v = 0; v < 8; ++v) acc2[n][v] = bv;
        }
#pragma unroll
        for (int kt = 0; kt < 4; ++kt) {
          v16h a;
          int kb = kt * 32 + ((lane & 16) ? 8 : 0);
#pragma unroll
          for (int e = 0; e < 8; ++e) {
            a[e]     = t_s[wv][arow][kb + e];
            a[e + 8] = t_s[wv][arow][kb + 16 + e];
          }
#pragma unroll
          for (int n = 0; n < 8; ++n) {
            v16h bf = *(const v16h*)(wB + (size_t)((n * 4 + kt) * 32 + lane) * 16);
            acc2[n] = __builtin_amdgcn_wmma_f32_16x16x32_f16(
                false, a, false, bf, (short)0, acc2[n], false, false);
          }
        }
        __syncthreads();
#pragma unroll
        for (int n = 0; n < 8; ++n) {
          int col = n * 16 + arow;
          float gv = gateL[col];
#pragma unroll
          for (int v = 0; v < 8; ++v)
            h_s[wv][v + rsel][col] += acc2[n][v] * gv;
        }
        __syncthreads();
      }  // NB blocks

      // fin: out(16x4) = h @ fin_W.T + fin_b ; 2 dot products per lane
      {
        float o[2];
#pragma unroll
        for (int q = 0; q < 2; ++q) {
          int idx = lane * 2 + q;
          int r = idx >> 2, j = idx & 3;
          const float* fw = p.fin_W + ((size_t)i * 4 + j) * HDIM;
          float s = p.fin_b[i * 4 + j];
          for (int c = 0; c < HDIM; ++c) s += h_s[wv][r][c] * fw[c];
          o[q] = s;
        }
#pragma unroll
        for (int q = 0; q < 2; ++q) {
          int idx = lane * 2 + q;
          out_s[wv][idx >> 2][idx & 3] = o[q];
        }
      }
      __syncthreads();
      if (lane < 16) {
        float o0 = out_s[wv][lane][0], o1 = out_s[wv][lane][1];
        float o2 = out_s[wv][lane][2], o3 = out_s[wv][lane][3];
        s0 = 1.f / (1.f + expf(-(o0 + 2.f))) + 0.001f;
        s1 = 1.f / (1.f + expf(-(o2 + 2.f))) + 0.001f;
        zz_s[wv][lane][0] = (x_s[wv][lane][0] - o1) / s0;
        zz_s[wv][lane][1] = (x_s[wv][lane][1] - o3) / s1;
      }
      __syncthreads();
    }  // fixed-point iterations

    if (lane < 16) {
      z_s[wv][lane][0] = zz_s[wv][lane][0];
      z_s[wv][lane][1] = zz_s[wv][lane][1];
      lq_s[wv][lane] -= logf(s0) + logf(s1);   // scales from last _made evaluation
    }
    __syncthreads();
  }  // flow layers

  if (lane < 16) {
    float z0 = z_s[wv][lane][0], z1 = z_s[wv][lane][1];
    float lq = lq_s[wv][lane] - logf(6.283185307179586f) - 0.5f * (z0 * z0 + z1 * z1);
    atomicAdd(p.out, -lq * (1.f / (float)N_ROWS));
  }
}

// ---------------------------------------------------------------- launch
extern "C" void kernel_launch(void* const* d_in, const int* in_sizes, int n_in,
                              void* d_out, int out_size, void* d_ws, size_t ws_size,
                              hipStream_t stream) {
  auto F = [&](int k) { return (const float*)d_in[k]; };
  const float* samples = F(0);
  const float* x_diff  = F(1);
  const float* x_mt    = F(2);
  // workspace layout (bytes): ctx[0,32K) hc[32K,128K) gate[...] packW f16
  char* ws = (char*)d_ws;
  float*     ctx   = (float*)(ws + 0);        // 64*128*4        = 32768
  float*     hc    = (float*)(ws + 32768);    // 3*64*128*4      = 98304
  float*     gate  = (float*)(ws + 131072);   // 3*2*64*128*4    = 196608
  _Float16*  packW = (_Float16*)(ws + 327680);// 384*512*2       = 393216

  zero_kernel<<<1, 32, 0, stream>>>((float*)d_out);

  {  // diff branch: d_in[3..23], c_in=32 -> ctx cols 0..63
    CnnP c;
    c.x = x_diff;
    c.c1W = F(3);  c.c1b = F(4);  c.g1 = F(5);  c.b1 = F(6);  c.m1 = F(7);  c.v1 = F(8);
    c.c2W = F(9);  c.c2b = F(10); c.g2 = F(11); c.b2 = F(12); c.m2 = F(13); c.v2 = F(14);
    c.f1W = F(15); c.f1b = F(16); c.pw = F(17);
    c.g3 = F(18);  c.b3 = F(19);  c.m3 = F(20); c.v3 = F(21);
    c.f2W = F(22); c.f2b = F(23);
    c.ctx_out = ctx; c.c_in = 32; c.col_off = 0;
    cnn_kernel<<<NIMG, 160, 0, stream>>>(c);
  }
  {  // mt branch: d_in[24..44], c_in=16 -> ctx cols 64..127
    CnnP c;
    c.x = x_mt;
    c.c1W = F(24); c.c1b = F(25); c.g1 = F(26); c.b1 = F(27); c.m1 = F(28); c.v1 = F(29);
    c.c2W = F(30); c.c2b = F(31); c.g2 = F(32); c.b2 = F(33); c.m2 = F(34); c.v2 = F(35);
    c.f1W = F(36); c.f1b = F(37); c.pw = F(38);
    c.g3 = F(39);  c.b3 = F(40);  c.m3 = F(41); c.v3 = F(42);
    c.f2W = F(43); c.f2b = F(44);
    c.ctx_out = ctx; c.c_in = 16; c.col_off = 64;
    cnn_kernel<<<NIMG, 160, 0, stream>>>(c);
  }
  {  // hoisted ctx matmuls on the 64 unique context rows
    CtxmmP cp{ctx, F(47), F(48), F(46), F(53), F(54), hc, gate};
    ctxmm_kernel<<<(KF * 3 * NIMG * HDIM) / 256, 256, 0, stream>>>(cp);
  }
  {  // pack A_W / B_W into f16 WMMA B fragments
    PackP pp{F(49), F(51), packW};
    pack_kernel<<<(384 * 512) / 256, 256, 0, stream>>>(pp);
  }
  {  // fused flow: 32768 rows, 16 rows per wave, 4 waves per block
    FlowP fp;
    fp.samples = samples;
    fp.init_W = F(45); fp.A_b = F(50); fp.B_b = F(52);
    fp.fin_W = F(55);  fp.fin_b = F(56);
    fp.lu_lower = F(57); fp.lu_upper = F(58);
    fp.lu_udiag = F(59); fp.lu_bias = F(60);
    fp.perms = (const int*)d_in[61];
    fp.hc = hc; fp.gate = gate; fp.packW = packW;
    fp.out = (float*)d_out;
    flow_kernel<<<N_ROWS / (WPB * 16), WPB * 32, 0, stream>>>(fp);
  }
}
